// KernelSelfAttention_2525440770107
// MI455X (gfx1250) — compile-verified, matrix operand order checked by reading, not run on hardware
//
#include <hip/hip_runtime.h>

// ---------------- problem constants ----------------
#define BATCH   4
#define S_LEN   8192
#define HMODEL  768
#define NH      12
#define HD      64
#define KDIM    768          // GEMM K (= HMODEL)
#define NSTEPS  (KDIM/32)    // 24 k-steps of 32 for bf16 WMMA
#define BLK_M   64           // rows of hidden staged per block in k1
#define APITCH  776          // LDS row pitch in halves (768 + 8 pad -> bank spread)
#define M_TOTAL (BATCH*S_LEN)
#define NW_ELEM (HMODEL*HMODEL)

// ---------------- vector types for WMMA ----------------
typedef __attribute__((ext_vector_type(16))) __bf16 v16bf;
typedef __attribute__((ext_vector_type(8)))  float  v8f;

struct Frag {
    union {
        v16bf v;
        uint4 q[2];
        unsigned short us[16];
    };
};

// ---------------- helpers ----------------
__device__ __forceinline__ unsigned short f2bf(float x) {
    union { float f; unsigned int u; } c; c.f = x;
    unsigned int u = c.u;
    unsigned int r = (u + 0x7FFFu + ((u >> 16) & 1u)) >> 16;   // round-nearest-even
    return (unsigned short)r;
}
__device__ __forceinline__ float bf2f(unsigned short h) {
    union { unsigned int u; float f; } c; c.u = ((unsigned int)h) << 16;
    return c.f;
}
__device__ __forceinline__ unsigned int pack2(float a, float b) {
    return (unsigned int)f2bf(a) | ((unsigned int)f2bf(b) << 16);
}
__device__ __forceinline__ float elu1(float x) {          // elu(x)+1
    return x > 0.f ? x + 1.f : __expf(x);
}
__device__ __forceinline__ v8f wmma_bf16(const Frag& a, const Frag& b, v8f c) {
    return __builtin_amdgcn_wmma_f32_16x16x32_bf16(false, a.v, false, b.v,
                                                   (short)0, c, false, false);
}

// ================= kernel 0: W fp32->bf16, zero accumulators =================
__global__ __launch_bounds__(256)
void k0_prep(const float* __restrict__ Wq, const float* __restrict__ Wk,
             const float* __restrict__ Wv, unsigned short* __restrict__ Wb,
             float* __restrict__ kvF, float* __restrict__ ksum) {
    const int stride = gridDim.x * blockDim.x;
    for (int i = blockIdx.x * blockDim.x + threadIdx.x; i < NW_ELEM; i += stride) {
        Wb[i]             = f2bf(Wq[i]);
        Wb[NW_ELEM + i]   = f2bf(Wk[i]);
        Wb[2*NW_ELEM + i] = f2bf(Wv[i]);
    }
    for (int i = blockIdx.x * blockDim.x + threadIdx.x; i < BATCH*NH*HD*HD; i += stride)
        kvF[i] = 0.f;
    for (int i = blockIdx.x * blockDim.x + threadIdx.x; i < BATCH*NH*HD; i += stride)
        ksum[i] = 0.f;
}

// ================= kernel 1: fused QKV projections (bf16 WMMA) =================
// Q -> [bh, s, d] bf16 (normal),  K,V -> [bh, d, s] bf16 (transposed)
__global__ __launch_bounds__(256)
void k1_proj(const float* __restrict__ hid, const float* __restrict__ mask,
             const unsigned short* __restrict__ Wb,
             const float* __restrict__ bq, const float* __restrict__ bk,
             const float* __restrict__ bv,
             unsigned short* __restrict__ Qf, unsigned short* __restrict__ Kt,
             unsigned short* __restrict__ Vt) {
    extern __shared__ unsigned short atile[];        // BLK_M x APITCH bf16
    __shared__ float smask[BLK_M];
    __shared__ float sscr[8][272];                   // per-wave 16x17 f32 transpose pad

    const int tid = threadIdx.x;
    const int gm0 = blockIdx.x * BLK_M;              // first global row (b*S+s)

    if (tid < BLK_M) smask[tid] = mask[gm0 + tid];

    // stage A block fp32 -> bf16 into LDS
    for (int i = tid; i < BLK_M * (KDIM / 4); i += 256) {
        const int row = i / (KDIM / 4);
        const int c4  = i % (KDIM / 4);
        const float4 f = *(const float4*)&hid[(size_t)(gm0 + row) * KDIM + c4 * 4];
        unsigned short* dst = &atile[row * APITCH + c4 * 4];
        dst[0] = f2bf(f.x); dst[1] = f2bf(f.y); dst[2] = f2bf(f.z); dst[3] = f2bf(f.w);
    }
    __syncthreads();

    const int wave = tid >> 5, lane = tid & 31;
    const int ln = lane & 15, hi = lane >> 4;
    const int b   = gm0 / S_LEN;
    const int sb0 = gm0 % S_LEN;
    const v8f vzero = {0.f,0.f,0.f,0.f,0.f,0.f,0.f,0.f};

    for (int g = 0; g < 9; ++g) {                     // 3 proj x 3 n-pair groups
        const int proj = g / 3;
        const int pair = (g % 3) * 8 + wave;          // 0..23
        const int n0   = pair * 2;                    // first 16-col tile index
        const float* bias = (proj == 0) ? bq : ((proj == 1) ? bk : bv);
        const unsigned short* Wp = Wb + (size_t)proj * NW_ELEM;

        v8f acc[4][2];
        #pragma unroll
        for (int ms = 0; ms < 4; ++ms)
            { acc[ms][0] = vzero; acc[ms][1] = vzero; }

        for (int k = 0; k < NSTEPS; ++k) {
            const int off1 = k * 32 + hi * 8;         // half-index of first 8-chunk
            Frag af[4], bfb[2];
            #pragma unroll
            for (int ms = 0; ms < 4; ++ms) {
                const unsigned short* p = &atile[(ms * 16 + ln) * APITCH + off1];
                af[ms].q[0] = *(const uint4*)p;
                af[ms].q[1] = *(const uint4*)(p + 16);
            }
            #pragma unroll
            for (int j = 0; j < 2; ++j) {
                const unsigned short* p =
                    &Wp[(size_t)((n0 + j) * 16 + ln) * KDIM + off1];
                bfb[j].q[0] = *(const uint4*)p;
                bfb[j].q[1] = *(const uint4*)(p + 16);
            }
            #pragma unroll
            for (int ms = 0; ms < 4; ++ms)
                #pragma unroll
                for (int j = 0; j < 2; ++j)
                    acc[ms][j] = wmma_bf16(af[ms], bfb[j], acc[ms][j]);
        }

        // ---- epilogue ----
        #pragma unroll
        for (int ms = 0; ms < 4; ++ms) {
            #pragma unroll
            for (int j = 0; j < 2; ++j) {
                const int ng = (n0 + j) * 16 + ln;    // lane's output column
                const int h  = ng / HD, d = ng % HD;
                const float bb = bias[ng];
                const int sb = sb0 + ms * 16 + hi * 8; // first of lane's 8 rows

                if (proj == 2) {                       // V: bias, transposed store
                    float vv[8];
                    #pragma unroll
                    for (int r = 0; r < 8; ++r) vv[r] = acc[ms][j][r] + bb;
                    uint4 w;
                    w.x = pack2(vv[0], vv[1]); w.y = pack2(vv[2], vv[3]);
                    w.z = pack2(vv[4], vv[5]); w.w = pack2(vv[6], vv[7]);
                    *(uint4*)&Vt[((size_t)(b * NH + h) * HD + d) * S_LEN + sb] = w;
                } else if (proj == 1) {                // K: mask+elu1, transposed
                    float vv[8];
                    #pragma unroll
                    for (int r = 0; r < 8; ++r)
                        vv[r] = elu1(acc[ms][j][r] + bb + smask[ms * 16 + hi * 8 + r]);
                    uint4 w;
                    w.x = pack2(vv[0], vv[1]); w.y = pack2(vv[2], vv[3]);
                    w.z = pack2(vv[4], vv[5]); w.w = pack2(vv[6], vv[7]);
                    *(uint4*)&Kt[((size_t)(b * NH + h) * HD + d) * S_LEN + sb] = w;
                } else {                               // Q: mask+elu1, LDS transpose
                    __builtin_amdgcn_wave_barrier();
                    #pragma unroll
                    for (int r = 0; r < 8; ++r)
                        sscr[wave][(hi * 8 + r) * 17 + ln] =
                            elu1(acc[ms][j][r] + bb + smask[ms * 16 + hi * 8 + r]);
                    __builtin_amdgcn_wave_barrier();
                    const float* src = &sscr[wave][ln * 17 + hi * 8];
                    uint4 w;
                    w.x = pack2(src[0], src[1]); w.y = pack2(src[2], src[3]);
                    w.z = pack2(src[4], src[5]); w.w = pack2(src[6], src[7]);
                    const int sQ    = sb0 + ms * 16 + ln;
                    const int dbase = (((n0 + j) * 16) % HD) + hi * 8;
                    *(uint4*)&Qf[((size_t)(b * NH + h) * S_LEN + sQ) * HD + dbase] = w;
                    __builtin_amdgcn_wave_barrier();
                }
            }
        }
    }
}

// ================= kernel 2: kv = K^T V and ksum = sum_s K =================
__global__ __launch_bounds__(256)
void k2_kv(const unsigned short* __restrict__ Kt, const unsigned short* __restrict__ Vt,
           float* __restrict__ kvF, float* __restrict__ ksum) {
    const int bh = blockIdx.x >> 4;
    const int s0 = (blockIdx.x & 15) * (S_LEN / 16);  // 512-length slice
    const int tid = threadIdx.x;

    // ---- ksum partial (row sums of Kt over this slice) ----
    {
        const int d = tid >> 2, part = tid & 3;
        const unsigned short* p = &Kt[((size_t)bh * HD + d) * S_LEN + s0 + part * 128];
        float sum = 0.f;
        for (int i = 0; i < 16; ++i) {
            uint4 qv = *(const uint4*)(p + i * 8);
            const unsigned short* u = (const unsigned short*)&qv;
            #pragma unroll
            for (int e = 0; e < 8; ++e) sum += bf2f(u[e]);
        }
        atomicAdd(&ksum[bh * HD + d], sum);
    }

    // ---- kv tiles via WMMA: A = Kt[d,s], B^T = Vt[e,s] ----
    const int wave = tid >> 5, lane = tid & 31, ln = lane & 15, hi = lane >> 4;
    const int dt = wave >> 1;                 // d tile 0..3
    const int e0 = (wave & 1) * 2;            // e tiles e0, e0+1
    const v8f vzero = {0.f,0.f,0.f,0.f,0.f,0.f,0.f,0.f};
    v8f acc[2] = {vzero, vzero};

    const unsigned short* Ar = &Kt[((size_t)bh * HD + dt * 16 + ln) * S_LEN + s0];
    const unsigned short* B0 = &Vt[((size_t)bh * HD + (e0    ) * 16 + ln) * S_LEN + s0];
    const unsigned short* B1 = &Vt[((size_t)bh * HD + (e0 + 1) * 16 + ln) * S_LEN + s0];

    for (int k = 0; k < 16; ++k) {
        const int off1 = k * 32 + hi * 8;
        Frag a, b0, b1;
        a.q[0]  = *(const uint4*)(Ar + off1); a.q[1]  = *(const uint4*)(Ar + off1 + 16);
        b0.q[0] = *(const uint4*)(B0 + off1); b0.q[1] = *(const uint4*)(B0 + off1 + 16);
        b1.q[0] = *(const uint4*)(B1 + off1); b1.q[1] = *(const uint4*)(B1 + off1 + 16);
        acc[0] = wmma_bf16(a, b0, acc[0]);
        acc[1] = wmma_bf16(a, b1, acc[1]);
    }
    #pragma unroll
    for (int j = 0; j < 2; ++j) {
        const int e = (e0 + j) * 16 + ln;
        #pragma unroll
        for (int r = 0; r < 8; ++r) {
            const int d = dt * 16 + hi * 8 + r;
            atomicAdd(&kvF[(size_t)bh * (HD * HD) + e * HD + d], acc[j][r]);
        }
    }
}

// ================= kernel 2b: kv f32 -> bf16 (stored as kv^T already) =========
__global__ __launch_bounds__(256)
void k2b_cvt(const float* __restrict__ kvF, unsigned short* __restrict__ kvT) {
    const int i = blockIdx.x * 256 + threadIdx.x;
    if (i < BATCH * NH * HD * HD) kvT[i] = f2bf(kvF[i]);
}

// ================= kernel 3: out = (Q kv) / (Q ksum) =================
__global__ __launch_bounds__(256)
void k3_out(const unsigned short* __restrict__ Qf, const unsigned short* __restrict__ kvT,
            const float* __restrict__ ksum, float* __restrict__ out) {
    __shared__ float sks[HD];
    const int tid = threadIdx.x;
    const int bh  = blockIdx.x >> 6;
    const int blk = blockIdx.x & 63;
    if (tid < HD) sks[tid] = ksum[bh * HD + tid];
    __syncthreads();

    const int wave = tid >> 5, lane = tid & 31, ln = lane & 15, hi = lane >> 4;
    const int st = blk * 8 + wave;                  // s-tile 0..511
    const int s_row = st * 16 + ln;

    const unsigned short* Ar = &Qf[((size_t)bh * S_LEN + s_row) * HD];
    const v8f vzero = {0.f,0.f,0.f,0.f,0.f,0.f,0.f,0.f};
    v8f acc[4] = {vzero, vzero, vzero, vzero};
    float part = 0.f;

    #pragma unroll
    for (int k = 0; k < 2; ++k) {
        const int off1 = k * 32 + hi * 8;
        Frag a;
        a.q[0] = *(const uint4*)(Ar + off1);
        a.q[1] = *(const uint4*)(Ar + off1 + 16);
        // denominator partial: lane holds Q[s_row, off1..off1+7] and [+16..+23]
        #pragma unroll
        for (int e = 0; e < 8; ++e) part += bf2f(a.us[e])     * sks[off1 + e];
        #pragma unroll
        for (int e = 0; e < 8; ++e) part += bf2f(a.us[8 + e]) * sks[off1 + 16 + e];
        #pragma unroll
        for (int j = 0; j < 4; ++j) {
            const unsigned short* Br = &kvT[(size_t)bh * (HD * HD) + (j * 16 + ln) * HD];
            Frag bfr;
            bfr.q[0] = *(const uint4*)(Br + off1);
            bfr.q[1] = *(const uint4*)(Br + off1 + 16);
            acc[j] = wmma_bf16(a, bfr, acc[j]);
        }
    }

    const float dsum = part + __shfl_xor(part, 16, 32);   // combine K-halves
    float dr[8];
    #pragma unroll
    for (int r = 0; r < 8; ++r) dr[r] = __shfl(dsum, r + hi * 8, 32);

    const int b_ = bh / NH, h_ = bh % NH;
    #pragma unroll
    for (int j = 0; j < 4; ++j)
        #pragma unroll
        for (int r = 0; r < 8; ++r) {
            const int s = st * 16 + hi * 8 + r;
            out[((size_t)b_ * S_LEN + s) * HMODEL + h_ * HD + j * 16 + ln] =
                acc[j][r] / dr[r];
        }
}

// ================= launch =================
extern "C" void kernel_launch(void* const* d_in, const int* in_sizes, int n_in,
                              void* d_out, int out_size, void* d_ws, size_t ws_size,
                              hipStream_t stream) {
    const float* hid  = (const float*)d_in[0];
    const float* mask = (const float*)d_in[1];
    const float* Wq   = (const float*)d_in[2];
    const float* bq   = (const float*)d_in[3];
    const float* Wk   = (const float*)d_in[4];
    const float* bk   = (const float*)d_in[5];
    const float* Wv   = (const float*)d_in[6];
    const float* bv   = (const float*)d_in[7];
    float* out = (float*)d_out;

    char* w = (char*)d_ws;                       // 256B-aligned offsets
    unsigned short* Wb  = (unsigned short*)(w);                 // 3.38 MB
    unsigned short* Qf  = (unsigned short*)(w + 3538944);       // 48 MB
    unsigned short* Kt  = (unsigned short*)(w + 53870592);      // 48 MB
    unsigned short* Vt  = (unsigned short*)(w + 104202240);     // 48 MB
    float*          kvF = (float*)(w + 154533888);              // 768 KB
    unsigned short* kvT = (unsigned short*)(w + 155320320);     // 384 KB
    float*          ks  = (float*)(w + 155713536);              // 12 KB

    k0_prep<<<1024, 256, 0, stream>>>(Wq, Wk, Wv, Wb, kvF, ks);
    k1_proj<<<M_TOTAL / BLK_M, 256, BLK_M * APITCH * 2, stream>>>(
        hid, mask, Wb, bq, bk, bv, Qf, Kt, Vt);
    k2_kv<<<BATCH * NH * 16, 256, 0, stream>>>(Kt, Vt, kvF, ks);
    k2b_cvt<<<(BATCH * NH * HD * HD) / 256, 256, 0, stream>>>(kvF, kvT);
    k3_out<<<(BATCH * NH * (S_LEN / 16)) / 8, 256, 0, stream>>>(Qf, kvT, ks, out);
}
